// StandaloneAlignment_77326591197382
// MI455X (gfx1250) — matrix operand order repaired and checked by reading, not compiled, not run on hardware
//
#include <hip/hip_runtime.h>

typedef __attribute__((ext_vector_type(16))) _Float16 v16h;
typedef __attribute__((ext_vector_type(8)))  float    v8f;

union AFrag { v16h v; _Float16 e[16]; };
union CFrag { v8f  v; float    e[8];  };

__device__ __forceinline__ void cvt4(_Float16* d, float4 v) {
  d[0] = (_Float16)v.x; d[1] = (_Float16)v.y;
  d[2] = (_Float16)v.z; d[3] = (_Float16)v.w;
}

// ---------------------------------------------------------------------------
// WMMA GEMM:  C[m,n] = sum_k A[m,k] * B[k,n]  (+bias, +GELU or +residual)
// A: (M,K) row-major f32 (per batch, stride sA), rows 16B-aligned
// W: (K,N) row-major f32 (stride sB; 0 for shared weights)
// One wave -> one 16x16 tile, 8 waves / block along M. M,N multiples of 16.
// K loop: guard-free float4 loads for full 32-chunks. Remainder chunk uses the
// SAME unguarded vector loads (caller guarantees the <=27-float over-read past
// K stays inside the workspace / adjacent rows) + f16 zero-selects, so no
// per-element address math and no exec-mask divergence.
// EPI: 0 = bias only (bias may be null), 1 = bias+exact GELU, 2 = bias+residual
// ---------------------------------------------------------------------------
template <int EPI>
__global__ __launch_bounds__(256) void gemm_wmma_kernel(
    const float* __restrict__ A, const float* __restrict__ W,
    const float* __restrict__ bias, const float* __restrict__ res,
    float* __restrict__ Cout,
    int M, int N, int K, int lda, int ldb, int ldc,
    long long sA, long long sB, long long sC)
{
  const int lane = threadIdx.x & 31;
  const int wave = threadIdx.x >> 5;
  const int b    = blockIdx.z;
  const int tile_n = blockIdx.x * 16;
  const int tile_m = (blockIdx.y * 8 + wave) * 16;
  if (tile_m >= M) return;

  const int hi  = lane >> 4;   // 0 or 1
  const int l15 = lane & 15;

  const float* Arow = A + (long long)b * sA + (long long)(tile_m + l15) * lda;
  const float* Wb   = W + (long long)b * sB;

  CFrag acc;
  #pragma unroll
  for (int i = 0; i < 8; ++i) acc.e[i] = 0.f;

  const int kfull = K & ~31;

  for (int k0 = 0; k0 < kfull; k0 += 32) {
    // A fragment (ISA 16-bit A 16x32): lane row = l15,
    // elems 0..7 -> k = k0+hi*8+i ; elems 8..15 -> k = k0+16+hi*8+i
    const float* ap = Arow + k0 + hi * 8;
    float4 a0 = *(const float4*)(ap);
    float4 a1 = *(const float4*)(ap + 4);
    float4 a2 = *(const float4*)(ap + 16);
    float4 a3 = *(const float4*)(ap + 20);
    // B fragment (ISA B 32x16): lane col n = tile_n+l15, elems k same as A
    const float* wp = Wb + (long long)(k0 + hi * 8) * ldb + tile_n + l15;
    _Float16 bz[16];
    #pragma unroll
    for (int i = 0; i < 8; ++i) {
      bz[i]     = (_Float16)wp[(long long)i * ldb];
      bz[8 + i] = (_Float16)wp[(long long)(16 + i) * ldb];
    }
    __builtin_prefetch(ap + 32, 0, 1);   // speculative; dropped if invalid

    AFrag a, bm;
    cvt4(a.e,      a0); cvt4(a.e +  4, a1);
    cvt4(a.e +  8, a2); cvt4(a.e + 12, a3);
    #pragma unroll
    for (int i = 0; i < 16; ++i) bm.e[i] = bz[i];

    acc.v = __builtin_amdgcn_wmma_f32_16x16x32_f16(
        false, a.v, false, bm.v, (short)0, acc.v, false, false);
  }

  if (kfull < K) {  // remainder chunk: unguarded vector loads + zero selects
    const float* ap = Arow + kfull + hi * 8;
    float4 a0 = *(const float4*)(ap);
    float4 a1 = *(const float4*)(ap + 4);
    float4 a2 = *(const float4*)(ap + 16);
    float4 a3 = *(const float4*)(ap + 20);
    const float* wp = Wb + tile_n + l15;
    _Float16 bw[16];
    #pragma unroll
    for (int i = 0; i < 8; ++i) {
      const int k1 = kfull + hi * 8 + i, k2 = k1 + 16;
      bw[i]     = (k1 < K) ? (_Float16)wp[(long long)k1 * ldb] : (_Float16)0.f;
      bw[8 + i] = (k2 < K) ? (_Float16)wp[(long long)k2 * ldb] : (_Float16)0.f;
    }

    _Float16 az[16], bz[16];
    cvt4(az, a0); cvt4(az + 4, a1); cvt4(az + 8, a2); cvt4(az + 12, a3);
    for (int i = 0; i < 16; ++i) bz[i] = bw[i];

    AFrag a, bm;
    const int ka = kfull + hi * 8;
    #pragma unroll
    for (int i = 0; i < 8; ++i) {
      a.e[i]     = (ka + i      < K) ? az[i]     : (_Float16)0.f;
      a.e[8 + i] = (ka + 16 + i < K) ? az[8 + i] : (_Float16)0.f;
    }
    #pragma unroll
    for (int i = 0; i < 16; ++i) bm.e[i] = bz[i];

    acc.v = __builtin_amdgcn_wmma_f32_16x16x32_f16(
        false, a.v, false, bm.v, (short)0, acc.v, false, false);
  }

  // Epilogue. C/D layout: n = lane&15, m = tile_m + v + hi*8.
  const int n = tile_n + l15;
  const float bval = bias ? bias[n] : 0.f;
  float* Cb = Cout + (long long)b * sC;
  const float* Rb = res ? (res + (long long)b * sC) : nullptr;
  #pragma unroll
  for (int v = 0; v < 8; ++v) {
    int m = tile_m + v + hi * 8;
    float val = acc.e[v] + bval;
    if (EPI == 1) val = 0.5f * val * (1.f + erff(val * 0.70710678118654752f));
    if (EPI == 2) val += Rb[(long long)m * ldc + n];
    Cb[(long long)m * ldc + n] = val;
  }
}

// ---------------------------------------------------------------------------
// (B,C,T) -> (B,T,C) transpose
// ---------------------------------------------------------------------------
__global__ void transpose_bct(const float* __restrict__ src, float* __restrict__ dst,
                              int Bn, int Cn, int T)
{
  long long idx = (long long)blockIdx.x * 256 + threadIdx.x;
  long long total = (long long)Bn * Cn * T;
  if (idx >= total) return;
  int c = (int)(idx % Cn);
  long long r = idx / Cn;
  int t = (int)(r % T);
  int b = (int)(r / T);
  dst[idx] = src[((long long)b * Cn + c) * T + t];
}

// src (R, Ccols) -> dst (Ccols, R)  (small weight transposes)
__global__ void transpose2d(const float* __restrict__ src, float* __restrict__ dst,
                            int R, int Ccols)
{
  int idx = blockIdx.x * 256 + threadIdx.x;
  if (idx >= R * Ccols) return;
  int r = idx / Ccols, c = idx % Ccols;
  dst[c * R + r] = src[idx];
}

// depthwise conv1d, kernel 7, pad 3, channels-last (B,T,C)
__global__ void dwconv7(const float* __restrict__ x, const float* __restrict__ w,
                        const float* __restrict__ bias, float* __restrict__ y,
                        int Bn, int T, int Cn)
{
  long long idx = (long long)blockIdx.x * 256 + threadIdx.x;
  long long total = (long long)Bn * T * Cn;
  if (idx >= total) return;
  int c = (int)(idx % Cn);
  long long r = idx / Cn;
  int t = (int)(r % T);
  int b = (int)(r / T);
  float acc = bias[c];
  #pragma unroll
  for (int j = 0; j < 7; ++j) {
    int tt = t + j - 3;
    if (tt >= 0 && tt < T)
      acc += w[c * 7 + j] * x[((long long)b * T + tt) * Cn + c];
  }
  y[idx] = acc;
}

// LayerNorm over C=80, one wave32 per row (lane covers c, c+32, c+64)
__global__ __launch_bounds__(256) void layernorm_c(
    const float* __restrict__ x, const float* __restrict__ g,
    const float* __restrict__ be, float* __restrict__ y, int rows, int Cn)
{
  int lane = threadIdx.x & 31;
  int wave = threadIdx.x >> 5;
  int row  = blockIdx.x * 8 + wave;
  if (row >= rows) return;
  const float* xr = x + (long long)row * Cn;
  float v0 = (lane      < Cn) ? xr[lane]      : 0.f;
  float v1 = (lane + 32 < Cn) ? xr[lane + 32] : 0.f;
  float v2 = (lane + 64 < Cn) ? xr[lane + 64] : 0.f;
  float s = v0 + v1 + v2;
  #pragma unroll
  for (int m = 16; m >= 1; m >>= 1) s += __shfl_xor(s, m, 32);
  float mean = s / (float)Cn;
  float d0 = v0 - mean, d1 = v1 - mean, d2 = v2 - mean;
  float q = ((lane      < Cn) ? d0 * d0 : 0.f) +
            ((lane + 32 < Cn) ? d1 * d1 : 0.f) +
            ((lane + 64 < Cn) ? d2 * d2 : 0.f);
  #pragma unroll
  for (int m = 16; m >= 1; m >>= 1) q += __shfl_xor(q, m, 32);
  float rstd = rsqrtf(q / (float)Cn + 1e-6f);
  float* yr = y + (long long)row * Cn;
  if (lane      < Cn) yr[lane]      = d0 * rstd * g[lane]      + be[lane];
  if (lane + 32 < Cn) yr[lane + 32] = d1 * rstd * g[lane + 32] + be[lane + 32];
  if (lane + 64 < Cn) yr[lane + 64] = d2 * rstd * g[lane + 64] + be[lane + 64];
}

// GRN part 1: Gx[b,h] = sqrt(sum_t h[b,t,h]^2)
__global__ void grn_gx(const float* __restrict__ h, float* __restrict__ Gx,
                       int Bn, int T, int H)
{
  int idx = blockIdx.x * 256 + threadIdx.x;
  if (idx >= Bn * H) return;
  int b = idx / H, hh = idx % H;
  const float* p = h + (long long)b * T * H + hh;
  float s = 0.f;
  for (int t = 0; t < T; ++t) { float v = p[(long long)t * H]; s += v * v; }
  Gx[idx] = sqrtf(s);
}

// GRN part 2: Nx[b,h] = Gx[b,h] / (mean_h Gx[b,:] + 1e-6)
__global__ __launch_bounds__(256) void grn_nx(const float* __restrict__ Gx,
                                              float* __restrict__ Nx, int H)
{
  __shared__ float sm[256];
  int b = blockIdx.x, tid = threadIdx.x;
  float v = (tid < H) ? Gx[b * H + tid] : 0.f;
  sm[tid] = v;
  __syncthreads();
  for (int s = 128; s > 0; s >>= 1) {
    if (tid < s) sm[tid] += sm[tid + s];
    __syncthreads();
  }
  float mean = sm[0] / (float)H;
  if (tid < H) Nx[b * H + tid] = v / (mean + 1e-6f);
}

// GRN part 3 (in-place): h = gamma*h*Nx + beta + h
__global__ void grn_apply(float* __restrict__ h, const float* __restrict__ Nx,
                          const float* __restrict__ gamma, const float* __restrict__ beta,
                          int Bn, int T, int H)
{
  long long idx = (long long)blockIdx.x * 256 + threadIdx.x;
  long long total = (long long)Bn * T * H;
  if (idx >= total) return;
  int hh = (int)(idx % H);
  int b  = (int)(idx / ((long long)T * H));
  float x = h[idx];
  h[idx] = gamma[hh] * x * Nx[b * H + hh] + beta[hh] + x;
}

__global__ void row_sqnorm(const float* __restrict__ x, float* __restrict__ out,
                           int rows, int Cn)
{
  int r = blockIdx.x * 256 + threadIdx.x;
  if (r >= rows) return;
  const float* p = x + (long long)r * Cn;
  float s = 0.f;
  for (int c = 0; c < Cn; ++c) s += p[c] * p[c];
  out[r] = s;
}

// In-place: S holds q.k dot products; convert to -(qn+kn-2dot)/C then softmax over T2
__global__ __launch_bounds__(256) void dist_softmax(
    float* __restrict__ S, const float* __restrict__ qn,
    const float* __restrict__ kn, int T1, int T2, int Cn)
{
  __shared__ float sm[256];
  int row = blockIdx.x;
  int t   = threadIdx.x;
  int b   = row / T1;
  float dot = S[(long long)row * T2 + t];
  float sc  = -(qn[row] + kn[b * T2 + t] - 2.f * dot) / (float)Cn;
  sm[t] = sc;
  __syncthreads();
  for (int s = 128; s > 0; s >>= 1) {
    if (t < s) sm[t] = fmaxf(sm[t], sm[t + s]);
    __syncthreads();
  }
  float mx = sm[0];
  __syncthreads();
  float e = expf(sc - mx);
  sm[t] = e;
  __syncthreads();
  for (int s = 128; s > 0; s >>= 1) {
    if (t < s) sm[t] += sm[t + s];
    __syncthreads();
  }
  S[(long long)row * T2 + t] = e / sm[0];
}

// ---------------------------------------------------------------------------
// Host orchestration of one projection stack
// bp: 30 device pointers = 3 blocks x {dw_w,dw_b,ln_g,ln_b,w1,b1,gamma,beta,w2,b2}
// ---------------------------------------------------------------------------
static void run_proj(const float* x, int Cin, int T,
                     const float* in_w, const float* in_b,
                     const float* const* bp,
                     float* feat, float* bufT, float* wTb, float* t1b, float* t2b,
                     float* hbuf, float* Gx, float* Nx, hipStream_t stream)
{
  const int B = 4, C = 80, H = 240;
  long long tot;

  tot = (long long)B * Cin * T;
  transpose_bct<<<(int)((tot + 255) / 256), 256, 0, stream>>>(x, bufT, B, Cin, T);

  // in_w (C, Cin) -> wTb (Cin, C) so the GEMM B operand is (K,N) row-major
  transpose2d<<<(C * Cin + 255) / 256, 256, 0, stream>>>(in_w, wTb, C, Cin);

  gemm_wmma_kernel<0><<<dim3(C / 16, T / 128, B), 256, 0, stream>>>(
      bufT, wTb, in_b, nullptr, feat, T, C, Cin, Cin, C, C,
      (long long)T * Cin, 0LL, (long long)T * C);

  for (int i = 0; i < 3; ++i) {
    const float* const* p = bp + 10 * i;
    const float *dw_w = p[0], *dw_b = p[1], *ln_g = p[2], *ln_b = p[3];
    const float *w1 = p[4], *b1 = p[5], *gamma = p[6], *beta = p[7];
    const float *w2 = p[8], *b2 = p[9];

    tot = (long long)B * T * C;
    dwconv7<<<(int)((tot + 255) / 256), 256, 0, stream>>>(feat, dw_w, dw_b, t1b, B, T, C);

    layernorm_c<<<(B * T) / 8, 256, 0, stream>>>(t1b, ln_g, ln_b, t2b, B * T, C);

    gemm_wmma_kernel<1><<<dim3(H / 16, T / 128, B), 256, 0, stream>>>(
        t2b, w1, b1, nullptr, hbuf, T, H, C, C, H, H,
        (long long)T * C, 0LL, (long long)T * H);

    grn_gx<<<(B * H + 255) / 256, 256, 0, stream>>>(hbuf, Gx, B, T, H);
    grn_nx<<<B, 256, 0, stream>>>(Gx, Nx, H);

    tot = (long long)B * T * H;
    grn_apply<<<(int)((tot + 255) / 256), 256, 0, stream>>>(hbuf, Nx, gamma, beta, B, T, H);

    gemm_wmma_kernel<2><<<dim3(C / 16, T / 128, B), 256, 0, stream>>>(
        hbuf, w2, b2, feat, feat, T, C, H, H, C, C,
        (long long)T * H, 0LL, (long long)T * C);
  }
}

extern "C" void kernel_launch(void* const* d_in, const int* in_sizes, int n_in,
                              void* d_out, int out_size, void* d_ws, size_t ws_size,
                              hipStream_t stream)
{
  (void)in_sizes; (void)n_in; (void)out_size; (void)ws_size;
  const int B = 4, C = 80, T1 = 1024, T2 = 256, CIN_Q = 100, CIN_K = 512;

  auto P = [&](int i) { return (const float*)d_in[i]; };

  // Workspace layout (floats): ~10.8 MB, all offsets multiples of 64 floats.
  // Buffer order matters: GEMM remainder chunks may over-read <=27 floats past
  // the end of an A-matrix row; each A buffer is followed by more workspace.
  float* ws   = (float*)d_ws;
  float* bufT = ws;                              // 4*512*256 = 524288 (transpose scratch)
  float* qf   = bufT + 4 * 512 * 256;            // 4*1024*80
  float* kf   = qf   + 4 * 1024 * 80;            // 4*256*80
  float* t1b  = kf   + 4 * 256 * 80;             // 4*1024*80
  float* t2b  = t1b  + 4 * 1024 * 80;            // 4*1024*80
  float* hbuf = t2b  + 4 * 1024 * 80;            // 4*1024*240
  float* kT   = hbuf + 4 * 1024 * 240;           // 4*80*256 = 81920
  float* wTb  = kT   + 4 * 80 * 256;             // 512*80 = 40960
  float* Gx   = wTb  + 512 * 80;                 // 960
  float* Nx   = Gx   + 4 * 240;                  // 960
  float* qn   = Nx   + 4 * 240;                  // 4096
  float* kn   = qn   + 4 * 1024;                 // 1024

  // Param pointer tables (setup_inputs insertion order, recursively flattened):
  // 0:queries 1:keys  2:q.in_w 3:q.in_b  4..33:q.blocks  34:k.in_w 35:k.in_b  36..65:k.blocks
  const float* qblocks[30];
  const float* kblocks[30];
  for (int i = 0; i < 30; ++i) { qblocks[i] = P(4 + i); kblocks[i] = P(36 + i); }

  run_proj((const float*)d_in[0], CIN_Q, T1, P(2), P(3), qblocks,
           qf, bufT, wTb, t1b, t2b, hbuf, Gx, Nx, stream);
  run_proj((const float*)d_in[1], CIN_K, T2, P(34), P(35), kblocks,
           kf, bufT, wTb, t1b, t2b, hbuf, Gx, Nx, stream);

  // Distance attention: -(qn + kn - 2 q.k)/C, softmax over T2, in-place on d_out
  row_sqnorm<<<(B * T1 + 255) / 256, 256, 0, stream>>>(qf, qn, B * T1, C);
  row_sqnorm<<<(B * T2 + 255) / 256, 256, 0, stream>>>(kf, kn, B * T2, C);

  // kf (B,T2,C) -> kT (B,C,T2): GEMM B operand (K=C rows, N=T2 cols) row-major
  transpose_bct<<<(B * T2 * C + 255) / 256, 256, 0, stream>>>(kf, kT, B, T2, C);

  float* S = (float*)d_out;
  gemm_wmma_kernel<0><<<dim3(T2 / 16, T1 / 128, B), 256, 0, stream>>>(
      qf, kT, nullptr, nullptr, S, T1, T2, C, C, T2, T2,
      (long long)T1 * C, (long long)C * T2, (long long)T1 * T2);

  dist_softmax<<<B * T1, 256, 0, stream>>>(S, qn, kn, T1, T2, C);
}
